// GNN_19748259627218
// MI455X (gfx1250) — matrix operand order repaired and checked by reading, not compiled, not run on hardware
//
#include <hip/hip_runtime.h>
#include <hip/hip_bf16.h>

typedef __attribute__((ext_vector_type(16))) __bf16 v16bf;
typedef __attribute__((ext_vector_type(8)))  __bf16 v8bf;
typedef __attribute__((ext_vector_type(8)))  float  v8f;

#define GN 100000
#define GE 1600000
#define GD 128
#define GL 5
#define GNBOND 62
#define GEPS 1e-5f

// ---------------- utility kernels ----------------

__global__ void k_zero_f32(float* p, int n) {
    int t = blockIdx.x * blockDim.x + threadIdx.x;
    if (t < n) p[t] = 0.0f;
}

__global__ void k_fill_f32(float* p, float v, int n) {
    int t = blockIdx.x * blockDim.x + threadIdx.x;
    if (t < n) p[t] = v;
}

// deg[src[e]] += 1 for real edges (self loop contribution pre-filled as 1.0)
__global__ void k_degree(const int* __restrict__ ei, float* __restrict__ deg) {
    int e = blockIdx.x * blockDim.x + threadIdx.x;
    if (e < GE) atomicAdd(&deg[ei[e]], 1.0f);
}

__global__ void k_dinv(float* __restrict__ d) {
    int i = blockIdx.x * blockDim.x + threadIdx.x;
    if (i < GN) {
        float v = d[i];
        d[i] = (v > 0.0f) ? rsqrtf(v) : 0.0f;
    }
}

// norm[e] = dinv[src]*dinv[dst]; edges [E, E+N) are self loops
__global__ void k_norm(const int* __restrict__ ei, const float* __restrict__ dinv,
                       float* __restrict__ norm) {
    int e = blockIdx.x * blockDim.x + threadIdx.x;
    if (e >= GE + GN) return;
    int s, d;
    if (e < GE) { s = ei[e]; d = ei[GE + e]; }
    else        { s = e - GE; d = s; }
    norm[e] = dinv[s] * dinv[d];
}

// h_bf16[i, f] = bf16(x_emb[x[i], f])
__global__ void k_gather_emb(const int* __restrict__ x, const float* __restrict__ emb,
                             __bf16* __restrict__ hb) {
    int t = blockIdx.x * blockDim.x + threadIdx.x;
    if (t >= GN * GD) return;
    int i = t >> 7, f = t & (GD - 1);
    hb[t] = (__bf16)emb[x[i] * GD + f];
}

__global__ void k_cvt_weights(const float* __restrict__ w, __bf16* __restrict__ wb, int n) {
    int t = blockIdx.x * blockDim.x + threadIdx.x;
    if (t < n) wb[t] = (__bf16)w[t];
}

// ---------------- WMMA GEMM: hl = h_bf16 @ W^T + b ----------------
// One wave computes a 16-row tile for all 128 output columns.
// A (16x32 bf16): lane l holds row M=(l&15), K = kb + (l>=16 ? 16 : 0) + {0..15}
// B (32x16 bf16): lane l holds col N=(l&15), K = kb + (l>=16 ? 16 : 0) + {0..15};
//                 B[k,n] = W[n,k] -> 16 contiguous halves of W row n.
// C (16x16 f32):  VGPR r: lanes 0-15 -> M=r, lanes 16-31 -> M=8+r; N = lane&15.
__global__ void __launch_bounds__(256)
k_gemm_wmma(const __bf16* __restrict__ hb, const __bf16* __restrict__ Wl,
            const float* __restrict__ bias, float* __restrict__ hl) {
    const int wid  = (blockIdx.x * blockDim.x + threadIdx.x) >> 5;
    if (wid >= GN / 16) return;                       // wave-uniform exit
    const int lane  = threadIdx.x & 31;
    const int hilo  = (lane >> 4) & 1;
    const int l15   = lane & 15;
    const int mbase = wid * 16;

    // Load whole A row-block (16 x 128) once: 4 k-steps of v16bf.
    v16bf a[4];
    const __bf16* arow = hb + (mbase + l15) * GD;
#pragma unroll
    for (int kk = 0; kk < 4; ++kk) {
        int kb = kk * 32 + hilo * 16;
        *(v8bf*)&a[kk]       = *(const v8bf*)(arow + kb);
        *((v8bf*)&a[kk] + 1) = *(const v8bf*)(arow + kb + 8);
    }

#pragma unroll
    for (int nt = 0; nt < 8; ++nt) {
        const int col = nt * 16 + l15;
        const __bf16* brow = Wl + col * GD + hilo * 16;
        v8f c = {};
#pragma unroll
        for (int kk = 0; kk < 4; ++kk) {
            v16bf b = *(const v16bf*)(brow + kk * 32);
            c = __builtin_amdgcn_wmma_f32_16x16x32_bf16(
                    false, a[kk], false, b, (short)0, c, false, false);
        }
        const float bv = bias[col];
        const int row0 = mbase + hilo * 8;
#pragma unroll
        for (int r = 0; r < 8; ++r)
            hl[(row0 + r) * GD + col] = c[r] + bv;
    }
}

// ---------------- edge message + scatter-add ----------------
// One wave per edge; lane handles 4 contiguous features (float4 -> 512B/edge).
__global__ void k_scatter(const float* __restrict__ hl, const float* __restrict__ norm,
                          const int* __restrict__ ei, const int* __restrict__ ea,
                          const float* __restrict__ etab_l, float* __restrict__ agg) {
    long long t = (long long)blockIdx.x * blockDim.x + threadIdx.x;
    int e = (int)(t >> 5);
    if (e >= GE + GN) return;
    int lane = (int)(t & 31);
    int f = lane << 2;
    int s, d, eid;
    if (e < GE) { s = ei[e]; d = ei[GE + e]; eid = ea[e]; }
    else        { s = e - GE; d = s; eid = 0; }
    float nv = norm[e];
    float4 hv = *(const float4*)(hl + s * GD + f);
    float4 tv = *(const float4*)(etab_l + eid * GD + f);
    float* dst = agg + d * GD + f;
    atomicAdd(dst + 0, nv * (hv.x + tv.x));
    atomicAdd(dst + 1, nv * (hv.y + tv.y));
    atomicAdd(dst + 2, nv * (hv.z + tv.z));
    atomicAdd(dst + 3, nv * (hv.w + tv.w));
}

// ---------------- batch-norm column stats ----------------
// blockDim = 128 (one thread per column), grid-strided over rows.
__global__ void k_bn_stats(const float* __restrict__ agg, float* __restrict__ bnsum,
                           float* __restrict__ bnsq) {
    int col = threadIdx.x;
    float s = 0.0f, s2 = 0.0f;
    for (int i = blockIdx.x; i < GN; i += gridDim.x) {
        float v = agg[i * GD + col];
        s += v; s2 += v * v;
    }
    atomicAdd(&bnsum[col], s);
    atomicAdd(&bnsq[col], s2);
}

// ---------------- batch-norm apply (+ReLU), emit bf16 or final f32 ----------------
__global__ void k_bn_apply(const float* __restrict__ agg, const float* __restrict__ bnsum,
                           const float* __restrict__ bnsq, const float* __restrict__ gamma,
                           const float* __restrict__ beta, __bf16* __restrict__ hb,
                           float* __restrict__ out_f32, int do_relu, int write_f32) {
    int t = blockIdx.x * blockDim.x + threadIdx.x;
    if (t >= GN * GD) return;
    int f = t & (GD - 1);
    const float invN = 1.0f / (float)GN;
    float mean = bnsum[f] * invN;
    float var  = bnsq[f] * invN - mean * mean;
    float y = (agg[t] - mean) * rsqrtf(var + GEPS) * gamma[f] + beta[f];
    if (do_relu) y = fmaxf(y, 0.0f);
    if (write_f32) out_f32[t] = y;
    else           hb[t] = (__bf16)y;
}

// ---------------- launch ----------------

static inline size_t align256(size_t x) { return (x + 255) & ~(size_t)255; }

extern "C" void kernel_launch(void* const* d_in, const int* in_sizes, int n_in,
                              void* d_out, int out_size, void* d_ws, size_t ws_size,
                              hipStream_t stream) {
    const int*   x        = (const int*)d_in[0];
    const int*   ei       = (const int*)d_in[1];   // (2,E): [0..E)=src, [E..2E)=dst
    const int*   ea       = (const int*)d_in[2];
    const float* x_emb    = (const float*)d_in[3];
    const float* lin_w    = (const float*)d_in[4];
    const float* lin_b    = (const float*)d_in[5];
    const float* edge_tab = (const float*)d_in[6];
    const float* bn_gamma = (const float*)d_in[7];
    const float* bn_beta  = (const float*)d_in[8];
    float* agg = (float*)d_out;                    // reuse d_out as agg each layer

    // workspace layout
    char* ws = (char*)d_ws;
    size_t off = 0;
    float*  norm = (float*)(ws + off); off = align256(off + (size_t)(GE + GN) * 4);
    float*  dinv = (float*)(ws + off); off = align256(off + (size_t)GN * 4);
    __bf16* hb   = (__bf16*)(ws + off); off = align256(off + (size_t)GN * GD * 2);
    float*  hl   = (float*)(ws + off); off = align256(off + (size_t)GN * GD * 4);
    __bf16* Wb   = (__bf16*)(ws + off); off = align256(off + (size_t)GL * GD * GD * 2);
    float*  bnsum = (float*)(ws + off); off = align256(off + (size_t)GD * 4);
    float*  bnsq  = (float*)(ws + off); off = align256(off + (size_t)GD * 4);
    (void)ws_size; (void)in_sizes; (void)n_in; (void)out_size;

    const int B = 256;
    auto cdiv = [](long long a, long long b) { return (int)((a + b - 1) / b); };

    // degree -> dinv -> norm
    k_fill_f32<<<cdiv(GN, B), B, 0, stream>>>(dinv, 1.0f, GN);   // self-loop count
    k_degree<<<cdiv(GE, B), B, 0, stream>>>(ei, dinv);
    k_dinv<<<cdiv(GN, B), B, 0, stream>>>(dinv);
    k_norm<<<cdiv(GE + GN, B), B, 0, stream>>>(ei, dinv, norm);

    // h = bf16(x_emb[x]); weights -> bf16
    k_gather_emb<<<cdiv((long long)GN * GD, B), B, 0, stream>>>(x, x_emb, hb);
    k_cvt_weights<<<cdiv(GL * GD * GD, B), B, 0, stream>>>(lin_w, Wb, GL * GD * GD);

    const int nwaves = GN / 16;                       // 6250 waves, 8 per block
    const int gemm_blocks = cdiv(nwaves * 32, B);
    const long long scat_threads = (long long)(GE + GN) * 32;

    for (int l = 0; l < GL; ++l) {
        k_zero_f32<<<cdiv((long long)GN * GD, B), B, 0, stream>>>(agg, GN * GD);
        k_zero_f32<<<1, 256, 0, stream>>>(bnsum, GD);
        k_zero_f32<<<1, 256, 0, stream>>>(bnsq, GD);

        k_gemm_wmma<<<gemm_blocks, B, 0, stream>>>(
            hb, Wb + (size_t)l * GD * GD, lin_b + (size_t)l * GD, hl);

        k_scatter<<<cdiv(scat_threads, B), B, 0, stream>>>(
            hl, norm, ei, ea, edge_tab + (size_t)l * GNBOND * GD, agg);

        k_bn_stats<<<1024, GD, 0, stream>>>(agg, bnsum, bnsq);

        k_bn_apply<<<cdiv((long long)GN * GD, B), B, 0, stream>>>(
            agg, bnsum, bnsq, bn_gamma + (size_t)l * GD, bn_beta + (size_t)l * GD,
            hb, agg, /*do_relu=*/(l < GL - 1) ? 1 : 0, /*write_f32=*/(l == GL - 1) ? 1 : 0);
    }
}
